// GptInferenceModel_73040213835929
// MI455X (gfx1250) — matrix operand order repaired and checked by reading, not compiled, not run on hardware
//
#include <hip/hip_runtime.h>

// ---------------- model constants (from the reference) ----------------
constexpr int cB = 4, cV = 20000, cD = 512, cH = 8, cL = 4, cDFF = 2048;
constexpr int cP = 128, cCW = 192, cTOPK = 50, cSTEPS = 64, cDH = 64;

typedef __attribute__((ext_vector_type(16))) __bf16 v16bf;
typedef __attribute__((ext_vector_type(8)))  float  v8f;
typedef __attribute__((ext_vector_type(4)))  unsigned int u32x4;
typedef __attribute__((ext_vector_type(8)))  int i32x8;
typedef __attribute__((ext_vector_type(4)))  int i32x4;

#if defined(__HIP_DEVICE_COMPILE__) && __has_builtin(__builtin_amdgcn_tensor_load_to_lds)
#define HAS_TDM 1
#else
#define HAS_TDM 0
#endif

#if HAS_TDM
#if __has_builtin(__builtin_amdgcn_s_wait_tensorcnt)
#define WAITT(n) do { __builtin_amdgcn_s_wait_tensorcnt(n); asm volatile("" ::: "memory"); } while (0)
#else
#define WAITT(n) asm volatile("s_wait_tensorcnt %0" :: "n"(n) : "memory")
#endif
#endif

__device__ __forceinline__ unsigned short f2bf(float f) {
  unsigned u = __float_as_uint(f);
  unsigned r = ((u >> 16) & 1u) + 0x7FFFu;   // round-to-nearest-even
  return (unsigned short)((u + r) >> 16);
}

__device__ __forceinline__ unsigned rnghash(unsigned a, unsigned b) {
  unsigned x = a * 0x9E3779B9u ^ (b + 0x85EBCA6Bu);
  x ^= x >> 16; x *= 0x7FEB352Du; x ^= x >> 15; x *= 0x846CA68Bu; x ^= x >> 16;
  return x;
}

#if HAS_TDM
// Issue a TDM 2-D tile load: 64 rows x 32 bf16 elements, row stride = K elems.
// D# built per CDNA5 ISA (group0 128b, group1 256b).
__device__ __forceinline__ void tdm_issue_b(const unsigned short* gbase,
                                            long strideElems, int validRows,
                                            unsigned ldsByteOff) {
  unsigned long long ga = (unsigned long long)(size_t)gbase;
  u32x4 g0;
  g0.x = 1u;                                             // count=1, user mode
  g0.y = ldsByteOff;                                     // lds_addr
  g0.z = (unsigned)(ga & 0xFFFFFFFFu);                   // global_addr[31:0]
  g0.w = (unsigned)((ga >> 32) & 0x01FFFFFFu) | 0x80000000u; // addr[56:32] | type=2
  unsigned long long q0 = ((unsigned long long)1u << 16)           // data_size=1 (2 bytes)
                        | ((unsigned long long)32u << 48);         // tensor_dim0[15:0]=32
  unsigned long long q1 = ((unsigned long long)(unsigned)validRows << 16) // tensor_dim1
                        | ((unsigned long long)32u << 48);         // tile_dim0=32
  unsigned long long q2 = (unsigned long long)64u                  // tile_dim1=64
                        | ((unsigned long long)(unsigned)strideElems << 32); // dim0_stride[31:0]
  unsigned long long q3 = (unsigned long long)((strideElems >> 32) & 0xFFFF); // dim0_stride[47:32]
  i32x8 g1;
  g1[0] = (int)(q0 & 0xFFFFFFFFu); g1[1] = (int)(q0 >> 32);
  g1[2] = (int)(q1 & 0xFFFFFFFFu); g1[3] = (int)(q1 >> 32);
  g1[4] = (int)(q2 & 0xFFFFFFFFu); g1[5] = (int)(q2 >> 32);
  g1[6] = (int)(q3 & 0xFFFFFFFFu); g1[7] = (int)(q3 >> 32);
  i32x4 z4; for (int q = 0; q < 4; ++q) z4[q] = 0;
#if __clang_major__ >= 23
  i32x8 z8; for (int q = 0; q < 8; ++q) z8[q] = 0;
  __builtin_amdgcn_tensor_load_to_lds(g0, g1, z4, z4, z8, 0);
#else
  __builtin_amdgcn_tensor_load_to_lds(g0, g1, z4, z4, 0);
#endif
}
#endif

// ---------------- init: tokens + disallowed ----------------
__global__ void k_init(const int* __restrict__ prompt, int* __restrict__ tokens,
                       float* __restrict__ disallowed) {
  long i = (long)blockIdx.x * blockDim.x + threadIdx.x;
  if (i < (long)cB * cV) disallowed[i] = 0.0f;
  if (i < (long)cB * cCW) {
    int b = (int)(i / cCW), p = (int)(i % cCW);
    tokens[i] = (p < cP) ? prompt[b * cP + p] : 0;
  }
}

// ---------------- f32 -> bf16 conversions ----------------
__global__ void k_cvt(const float* __restrict__ s, unsigned short* __restrict__ d, long n) {
  long i = (long)blockIdx.x * blockDim.x + threadIdx.x;
  if (i < n) d[i] = f2bf(s[i]);
}
// W [K,N] f32 -> Wt [N,K] bf16
__global__ void k_cvt_t(const float* __restrict__ W, unsigned short* __restrict__ Wt,
                        int K, int N) {
  long i = (long)blockIdx.x * blockDim.x + threadIdx.x;
  if (i < (long)K * N) {
    int k = (int)(i / N), n = (int)(i % N);
    Wt[(size_t)n * K + k] = f2bf(W[i]);
  }
}

// ---------------- WMMA bf16 GEMM: C[M,N] = A[M,K] @ Bt[N,K]^T ----------------
// Block = 4 waves. Block owns one 16-row M strip (A staged in LDS as bf16,
// shared by all waves); each wave computes 4 N-tiles. B streamed per-wave in
// 64x32 chunks via TDM double-buffering in LDS (fallback: direct b128 loads).
__global__ __launch_bounds__(128) void k_gemm(
    const float* __restrict__ A, long lda,
    const unsigned short* __restrict__ Bt,
    float* __restrict__ C, long ldc,
    int M, int N, int K) {
  __shared__ unsigned short As[16 * 1024];          // 32 KB A panel (bf16)
#if HAS_TDM
  __shared__ unsigned short Bs[4 * 2 * 64 * 32];    // 32 KB: 4 waves x 2 bufs x (64x32)
#endif
  int NT = N >> 4;
  int MT = (M + 15) >> 4;
  int GN = (NT + 15) >> 4;
  int tm = blockIdx.x % MT;
  int g  = blockIdx.x / MT;
  if (g >= GN) return;
  int tid = threadIdx.x;
  int wv = tid >> 5, lane = tid & 31;
  int lr = lane & 15, hi = lane >> 4;
  int M0 = tm << 4;
  int t0 = g * 16 + wv * 4;                         // first of 4 N-tiles for this wave

  v8f acc[4];
  #pragma unroll
  for (int j = 0; j < 4; ++j)
    #pragma unroll
    for (int r = 0; r < 8; ++r) acc[j][r] = 0.0f;

#if HAS_TDM
  int nrowBase = ((t0 < NT) ? t0 : 0) << 4;         // 64 contiguous B rows
  int validRows = N - nrowBase;                     // TDM OOB-zeroes rows >= this
  unsigned myLds0 = (unsigned)(size_t)(&Bs[wv * 4096]);          // buffer 0
  unsigned myLds1 = (unsigned)(size_t)(&Bs[wv * 4096 + 2048]);   // buffer 1
#endif

  for (int kh = 0; kh < K; kh += 1024) {
    int KC = (K - kh < 1024) ? (K - kh) : 1024;
    int kcsh = 31 - __clz(KC);
    __syncthreads();                                // protect As re-stage
    // ---- stage A panel: f32 (strided) -> bf16 in LDS, pad rows zeroed ----
    for (int e = tid * 4; e < 16 * KC; e += 128 * 4) {
      int row = e >> kcsh;
      int k = e & (KC - 1);
      int m = M0 + row;
      int mc = (m < M) ? m : (M - 1);
      float sc = (m < M) ? 1.0f : 0.0f;
      float4 f4 = *(const float4*)(A + (size_t)mc * lda + kh + k);
      ushort4 o;
      o.x = f2bf(f4.x * sc); o.y = f2bf(f4.y * sc);
      o.z = f2bf(f4.z * sc); o.w = f2bf(f4.w * sc);
      *(ushort4*)(As + (size_t)row * 1024 + k) = o;
    }
    __syncthreads();

#if HAS_TDM
    if (lane == 0)
      tdm_issue_b(Bt + (size_t)nrowBase * K + kh, K, validRows, myLds0);
#endif
    for (int kk = 0; kk < KC; kk += 32) {
      // A fragment from LDS (16x32 bf16, contiguous k = 16*hi + s, matching B)
      v16bf af;
      const unsigned short* ap = As + (size_t)lr * 1024 + kk + hi * 16;
      ((uint4*)&af)[0] = *(const uint4*)(ap);
      ((uint4*)&af)[1] = *(const uint4*)(ap + 8);
#if HAS_TDM
      int cur = (kk >> 5) & 1;
      bool more = (kk + 32) < KC;
      if (lane == 0 && more)
        tdm_issue_b(Bt + (size_t)nrowBase * K + kh + kk + 32, K, validRows,
                    cur ? myLds0 : myLds1);
      if (more) WAITT(1); else WAITT(0);
      __builtin_amdgcn_wave_barrier();
      const unsigned short* bbase = Bs + wv * 4096 + cur * 2048;
      #pragma unroll
      for (int j = 0; j < 4; ++j) {
        v16bf bf;
        const unsigned short* bb = bbase + (j * 16 + lr) * 32 + hi * 16;
        ((uint4*)&bf)[0] = *(const uint4*)(bb);
        ((uint4*)&bf)[1] = *(const uint4*)(bb + 8);
        acc[j] = __builtin_amdgcn_wmma_f32_16x16x32_bf16(false, af, false, bf,
                                                         (short)0, acc[j], false, false);
      }
#else
      #pragma unroll
      for (int j = 0; j < 4; ++j) {
        int tj = t0 + j; if (tj >= NT) tj = NT - 1;
        const unsigned short* bb = Bt + ((size_t)(tj * 16 + lr)) * K + kh + kk + hi * 16;
        if (kk + 32 < KC) __builtin_prefetch(bb + 32, 0, 1);
        v16bf bf;
        ((uint4*)&bf)[0] = *(const uint4*)(bb);
        ((uint4*)&bf)[1] = *(const uint4*)(bb + 8);
        acc[j] = __builtin_amdgcn_wmma_f32_16x16x32_bf16(false, af, false, bf,
                                                         (short)0, acc[j], false, false);
      }
#endif
    }
  }
  // ---- store 4 16x16 tiles ----
  #pragma unroll
  for (int j = 0; j < 4; ++j) {
    int tj = t0 + j;
    if (tj < NT) {
      int n = (tj << 4) + lr;
      #pragma unroll
      for (int r = 0; r < 8; ++r) {
        int m = M0 + r + hi * 8;
        if (m < M) C[(size_t)m * ldc + n] = acc[j][r];
      }
    }
  }
}

// ---------------- attention (prefill Tq=P causal, decode Tq=1 over kvLen keys) ----------------
__global__ __launch_bounds__(128) void k_attn(
    const float* __restrict__ q, const float* __restrict__ kc, const float* __restrict__ vc,
    float* __restrict__ out, int Tq, int kvLen, long qStrideB, long oStrideB) {
  __shared__ float s[cCW];
  __shared__ float red[128];
  int gid = blockIdx.x;
  int p = gid % Tq;
  int h = (gid / Tq) % cH;
  int b = gid / (Tq * cH);
  int causal = (Tq == 1) ? (kvLen - 1) : p;
  const float* qv = q + (size_t)b * qStrideB + (size_t)p * cD + h * cDH;
  const float* kb = kc + (size_t)b * cCW * cD + h * cDH;
  const float* vb = vc + (size_t)b * cCW * cD + h * cDH;
  int t = threadIdx.x;
  for (int t0 = t; t0 < cCW; t0 += 128) {
    float sv = -1e9f;
    if (t0 < kvLen && t0 <= causal) {
      const float* kr = kb + (size_t)t0 * cD;
      float acc = 0.f;
      #pragma unroll 8
      for (int d = 0; d < cDH; ++d) acc += qv[d] * kr[d];
      sv = acc * 0.125f;                       // 1/sqrt(64)
    }
    s[t0] = sv;
  }
  __syncthreads();
  float lm = -1e9f;
  for (int t0 = t; t0 < cCW; t0 += 128) lm = fmaxf(lm, s[t0]);
  red[t] = lm; __syncthreads();
  for (int off = 64; off > 0; off >>= 1) { if (t < off) red[t] = fmaxf(red[t], red[t + off]); __syncthreads(); }
  float mx = red[0]; __syncthreads();
  float ls = 0.f;
  for (int t0 = t; t0 < cCW; t0 += 128) { float e = __expf(s[t0] - mx); s[t0] = e; ls += e; }
  red[t] = ls; __syncthreads();
  for (int off = 64; off > 0; off >>= 1) { if (t < off) red[t] += red[t + off]; __syncthreads(); }
  float inv = 1.f / red[0];
  __syncthreads();
  if (t < cDH) {
    float acc = 0.f;
    for (int t0 = 0; t0 < kvLen; ++t0) acc += s[t0] * vb[(size_t)t0 * cD + t];
    out[(size_t)b * oStrideB + (size_t)p * cD + h * cDH + t] = acc * inv;
  }
}

// ---------------- residual add + layernorm (eps 1e-6, no affine) ----------------
__global__ __launch_bounds__(256) void k_add_ln(
    const float* __restrict__ a, long aCW, long aPos,
    const float* __restrict__ c, long cCW, long cPos,
    float* o1, long o1CW, long o1Pos,
    float* o2, long o2CW, long o2Pos,
    int rowsPerB) {
  __shared__ float rs[256], rq[256];
  int row = blockIdx.x;
  int b = row / rowsPerB, p = row % rowsPerB;
  const float* ar = a + ((size_t)b * aCW + aPos + p) * cD;
  const float* cr = c + ((size_t)b * cCW + cPos + p) * cD;
  int t = threadIdx.x;
  float x0 = ar[t] + cr[t];
  float x1 = ar[t + 256] + cr[t + 256];
  rs[t] = x0 + x1;
  rq[t] = x0 * x0 + x1 * x1;
  __syncthreads();
  for (int off = 128; off > 0; off >>= 1) {
    if (t < off) { rs[t] += rs[t + off]; rq[t] += rq[t + off]; }
    __syncthreads();
  }
  float mean = rs[0] * (1.0f / cD);
  float var  = rq[0] * (1.0f / cD) - mean * mean;
  float rinv = rsqrtf(var + 1e-6f);
  float y0 = (x0 - mean) * rinv, y1 = (x1 - mean) * rinv;
  if (o1) { float* o = o1 + ((size_t)b * o1CW + o1Pos + p) * cD; o[t] = y0; o[t + 256] = y1; }
  if (o2) { float* o = o2 + ((size_t)b * o2CW + o2Pos + p) * cD; o[t] = y0; o[t + 256] = y1; }
}

// ---------------- gelu (tanh approximation, jax default) ----------------
__global__ void k_gelu(float* __restrict__ x, long n) {
  long i = (long)blockIdx.x * blockDim.x + threadIdx.x;
  if (i < n) {
    float v = x[i];
    float u = 0.7978845608028654f * (v + 0.044715f * v * v * v);
    x[i] = 0.5f * v * (1.0f + tanhf(u));
  }
}

// ---------------- embedding gathers ----------------
__global__ void k_embed_prefill(const int* __restrict__ tokens, const float* __restrict__ E,
                                float* __restrict__ emb) {
  int row = blockIdx.x;
  int b = row / cP, p = row % cP;
  int tok = tokens[b * cCW + p];
  const float* er = E + (size_t)tok * cD;
  float* o = emb + ((size_t)b * cCW + p) * cD;
  for (int d = threadIdx.x; d < cD; d += blockDim.x) o[d] = er[d];
}
__global__ void k_embed_row(const int* __restrict__ tokens, const float* __restrict__ E,
                            float* __restrict__ emb, int pos) {
  int b = blockIdx.x;
  int tok = tokens[b * cCW + pos - 1];
  const float* er = E + (size_t)tok * cD;
  float* o = emb + ((size_t)b * cCW + pos - 1) * cD;
  for (int d = threadIdx.x; d < cD; d += blockDim.x) o[d] = er[d];
}

// ---------------- block-recurring mask + write logits_seq ----------------
__global__ __launch_bounds__(256) void k_mask(
    const int* __restrict__ tokens, const float* __restrict__ logits,
    float* __restrict__ disallowed, float* __restrict__ outLogits,
    float* __restrict__ mwork, int pos) {
  __shared__ int rid[256];
  int b = blockIdx.x, t = threadIdx.x;
  int best = -1;
  for (int p = t; p < pos; p += 256)
    if (tokens[b * cCW + p] == 103) best = p;          // p increasing -> last sep
  rid[t] = best; __syncthreads();
  for (int off = 128; off > 0; off >>= 1) { if (t < off) rid[t] = max(rid[t], rid[t + off]); __syncthreads(); }
  int last = rid[0];
  int lastIdx = (last >= 0) ? last : (pos - 1);
  int newVisit = (lastIdx == pos - 1) ? 1 : 0;
  __syncthreads();
  for (int v = t; v < cV; v += 256) {
    size_t ix = (size_t)b * cV + v;
    float dd = newVisit ? 0.0f : disallowed[ix];
    disallowed[ix] = dd;
    float m = logits[ix] + dd * (-1e9f);
    outLogits[ix] = m;
    mwork[ix] = m;
  }
}

// ---------------- top-k(50) + categorical sample + next-token rule ----------------
__global__ __launch_bounds__(256) void k_topk_sample(
    float* __restrict__ mwork, float* __restrict__ disallowed,
    int* __restrict__ tokens, int pos, int step) {
  __shared__ float rv[256];
  __shared__ int   ri[256];
  __shared__ float topv[cTOPK];
  __shared__ int   topi[cTOPK];
  int b = blockIdx.x, t = threadIdx.x;
  for (int it = 0; it < cTOPK; ++it) {
    float bv = -3e38f; int bi = 0;
    for (int v = t; v < cV; v += 256) {
      float x = mwork[(size_t)b * cV + v];
      if (x > bv) { bv = x; bi = v; }
    }
    rv[t] = bv; ri[t] = bi; __syncthreads();
    for (int off = 128; off > 0; off >>= 1) {
      if (t < off) {
        if (rv[t + off] > rv[t] || (rv[t + off] == rv[t] && ri[t + off] < ri[t])) {
          rv[t] = rv[t + off]; ri[t] = ri[t + off];
        }
      }
      __syncthreads();
    }
    if (t == 0) {
      topv[it] = rv[0]; topi[it] = ri[0];
      mwork[(size_t)b * cV + ri[0]] = -3e38f;
    }
    __syncthreads();
  }
  if (t == 0) {
    float mx = topv[0], sum = 0.f, pr[cTOPK];
    for (int i = 0; i < cTOPK; ++i) { pr[i] = __expf(topv[i] - mx); sum += pr[i]; }
    unsigned h = rnghash((unsigned)(step + 42), (unsigned)b);
    float u = ((h >> 8) * (1.0f / 16777216.0f)) * sum;
    float cs = 0.f; int choice = cTOPK - 1;
    for (int i = 0; i < cTOPK; ++i) { cs += pr[i]; if (u <= cs) { choice = i; break; } }
    int sampled = topi[choice];
    disallowed[(size_t)b * cV + sampled] += 1.0f;
    int lastTok = tokens[b * cCW + pos - 1];
    int nxt = (lastTok >= 3 && lastTok < 103) ? 103 : sampled;  // ATT_IDS -> VS
    tokens[b * cCW + pos] = nxt;
  }
}

__global__ void k_out_tokens(const int* __restrict__ tokens, int* __restrict__ out) {
  int i = blockIdx.x * blockDim.x + threadIdx.x;
  if (i < cB * cCW) out[i] = tokens[i];
}

// ======================================================================
extern "C" void kernel_launch(void* const* d_in, const int* in_sizes, int n_in,
                              void* d_out, int out_size, void* d_ws, size_t ws_size,
                              hipStream_t stream) {
  (void)in_sizes; (void)n_in; (void)out_size; (void)ws_size;
  const int*   prompt = (const int*)d_in[0];
  const float* E  = (const float*)d_in[1];
  const float* Wq = (const float*)d_in[2];
  const float* Wk = (const float*)d_in[3];
  const float* Wv = (const float*)d_in[4];
  const float* Wo = (const float*)d_in[5];
  const float* W1 = (const float*)d_in[6];
  const float* W2 = (const float*)d_in[7];

  char* ws = (char*)d_ws;
  size_t off = 0;
  auto take = [&](size_t bytes) -> char* {
    char* p = ws + off;
    off = (off + bytes + 255) & ~(size_t)255;
    return p;
  };
  unsigned short* Ebf = (unsigned short*)take((size_t)cV * cD * 2);
  unsigned short* Wqt = (unsigned short*)take((size_t)cL * cD * cD * 2);
  unsigned short* Wkt = (unsigned short*)take((size_t)cL * cD * cD * 2);
  unsigned short* Wvt = (unsigned short*)take((size_t)cL * cD * cD * 2);
  unsigned short* Wot = (unsigned short*)take((size_t)cL * cD * cD * 2);
  unsigned short* W1t = (unsigned short*)take((size_t)cL * cD * cDFF * 2);
  unsigned short* W2t = (unsigned short*)take((size_t)cL * cDFF * cD * 2);
  float* embc   = (float*)take((size_t)cB * cCW * cD * 4);
  float* lcache = (float*)take((size_t)cL * cB * cCW * cD * 4);
  float* kcache = (float*)take((size_t)cL * cB * cCW * cD * 4);
  float* vcache = (float*)take((size_t)cL * cB * cCW * cD * 4);
  float* qtmp    = (float*)take((size_t)cB * cP * cD * 4);
  float* attnout = (float*)take((size_t)cB * cP * cD * 4);
  float* otmp    = (float*)take((size_t)cB * cP * cD * 4);
  float* x1tmp   = (float*)take((size_t)cB * cP * cD * 4);
  float* f2tmp   = (float*)take((size_t)cB * cP * cD * 4);
  float* fftmp   = (float*)take((size_t)cB * cP * cDFF * 4);
  float* logits     = (float*)take((size_t)cB * cV * 4);
  float* mwork      = (float*)take((size_t)cB * cV * 4);
  float* disallowed = (float*)take((size_t)cB * cV * 4);
  int*   tokens     = (int*)take((size_t)cB * cCW * 4);

  auto gemm = [&](const float* A, long lda, const unsigned short* Bt,
                  float* C, long ldc, int M, int N, int K) {
    int MT = (M + 15) / 16, NT = N / 16, GN = (NT + 15) / 16;
    k_gemm<<<MT * GN, 128, 0, stream>>>(A, lda, Bt, C, ldc, M, N, K);
  };

  // ---- init + precision conversion (weights -> transposed bf16) ----
  k_init<<<(cB * cV + 255) / 256, 256, 0, stream>>>(prompt, tokens, disallowed);
  long nE = (long)cV * cD;
  k_cvt<<<(int)((nE + 255) / 256), 256, 0, stream>>>(E, Ebf, nE);
  for (int l = 0; l < cL; ++l) {
    long nDD = (long)cD * cD, nDF = (long)cD * cDFF;
    k_cvt_t<<<(int)((nDD + 255) / 256), 256, 0, stream>>>(Wq + (size_t)l * nDD, Wqt + (size_t)l * nDD, cD, cD);
    k_cvt_t<<<(int)((nDD + 255) / 256), 256, 0, stream>>>(Wk + (size_t)l * nDD, Wkt + (size_t)l * nDD, cD, cD);
    k_cvt_t<<<(int)((nDD + 255) / 256), 256, 0, stream>>>(Wv + (size_t)l * nDD, Wvt + (size_t)l * nDD, cD, cD);
    k_cvt_t<<<(int)((nDD + 255) / 256), 256, 0, stream>>>(Wo + (size_t)l * nDD, Wot + (size_t)l * nDD, cD, cD);
    k_cvt_t<<<(int)((nDF + 255) / 256), 256, 0, stream>>>(W1 + (size_t)l * nDF, W1t + (size_t)l * nDF, cD, cDFF);
    k_cvt_t<<<(int)((nDF + 255) / 256), 256, 0, stream>>>(W2 + (size_t)l * nDF, W2t + (size_t)l * nDF, cDFF, cD);
  }

  // ---- prefill ----
  k_embed_prefill<<<cB * cP, 256, 0, stream>>>(tokens, E, embc);
  for (int i = 0; i < cL; ++i) {
    const float* inb = (i == 0) ? embc : lcache + (size_t)(i - 1) * cB * cCW * cD;
    for (int b = 0; b < cB; ++b) {
      const float* Ab = inb + (size_t)b * cCW * cD;
      gemm(Ab, cD, Wqt + (size_t)i * cD * cD, qtmp + (size_t)b * cP * cD, cD, cP, cD, cD);
      gemm(Ab, cD, Wkt + (size_t)i * cD * cD,
           kcache + ((size_t)(i * cB + b) * cCW) * cD, cD, cP, cD, cD);
      gemm(Ab, cD, Wvt + (size_t)i * cD * cD,
           vcache + ((size_t)(i * cB + b) * cCW) * cD, cD, cP, cD, cD);
    }
    k_attn<<<cB * cH * cP, 128, 0, stream>>>(qtmp,
        kcache + (size_t)i * cB * cCW * cD, vcache + (size_t)i * cB * cCW * cD,
        attnout, cP, cP, (long)cP * cD, (long)cP * cD);
    gemm(attnout, cD, Wot + (size_t)i * cD * cD, otmp, cD, cB * cP, cD, cD);
    k_add_ln<<<cB * cP, 256, 0, stream>>>(inb, cCW, 0, otmp, cP, 0,
                                          x1tmp, cP, 0, (float*)nullptr, 0, 0, cP);
    gemm(x1tmp, cD, W1t + (size_t)i * cD * cDFF, fftmp, cDFF, cB * cP, cDFF, cD);
    k_gelu<<<(int)(((long)cB * cP * cDFF + 255) / 256), 256, 0, stream>>>(fftmp, (long)cB * cP * cDFF);
    gemm(fftmp, cDFF, W2t + (size_t)i * cDFF * cD, f2tmp, cD, cB * cP, cD, cDFF);
    k_add_ln<<<cB * cP, 256, 0, stream>>>(x1tmp, cP, 0, f2tmp, cP, 0,
                                          (float*)nullptr, 0, 0,
                                          lcache + (size_t)i * cB * cCW * cD, cCW, 0, cP);
  }

  // ---- decode: 64 steps, KV-cached (output-equivalent to the reference recompute) ----
  for (int step = 0; step < cSTEPS; ++step) {
    int pos = cP + step;
    k_embed_row<<<cB, 256, 0, stream>>>(tokens, E, embc, pos);
    for (int i = 0; i < cL; ++i) {
      const float* ctxb = (i == 0) ? embc : lcache + (size_t)(i - 1) * cB * cCW * cD;
      const float* Arow = ctxb + (size_t)(pos - 1) * cD;   // row b at +b*CW*D
      long ldr = (long)cCW * cD;
      gemm(Arow, ldr, Wqt + (size_t)i * cD * cD, qtmp, cD, cB, cD, cD);
      gemm(Arow, ldr, Wkt + (size_t)i * cD * cD,
           kcache + ((size_t)i * cB * cCW + (pos - 1)) * cD, ldr, cB, cD, cD);
      gemm(Arow, ldr, Wvt + (size_t)i * cD * cD,
           vcache + ((size_t)i * cB * cCW + (pos - 1)) * cD, ldr, cB, cD, cD);
      k_attn<<<cB * cH, 128, 0, stream>>>(qtmp,
          kcache + (size_t)i * cB * cCW * cD, vcache + (size_t)i * cB * cCW * cD,
          attnout, 1, pos, (long)cD, (long)cD);
      gemm(attnout, cD, Wot + (size_t)i * cD * cD, otmp, cD, cB, cD, cD);
      k_add_ln<<<cB, 256, 0, stream>>>(ctxb, cCW, pos - 1, otmp, 1, 0,
                                       x1tmp, 1, 0, (float*)nullptr, 0, 0, 1);
      gemm(x1tmp, cD, W1t + (size_t)i * cD * cDFF, fftmp, cDFF, cB, cDFF, cD);
      k_gelu<<<(int)(((long)cB * cDFF + 255) / 256), 256, 0, stream>>>(fftmp, (long)cB * cDFF);
      gemm(fftmp, cDFF, W2t + (size_t)i * cDFF * cD, f2tmp, cD, cB, cD, cDFF);
      k_add_ln<<<cB, 256, 0, stream>>>(x1tmp, 1, 0, f2tmp, 1, 0,
                                       (float*)nullptr, 0, 0,
                                       lcache + (size_t)i * cB * cCW * cD, cCW, pos - 1, 1);
    }
    // tied-embedding logits: [B,512] x Ebf[V,512]^T  (WMMA, 1250 N-tiles)
    gemm(lcache + (size_t)(cL - 1) * cB * cCW * cD + (size_t)(pos - 1) * cD,
         (long)cCW * cD, Ebf, logits, cV, cB, cV, cD);
    float* outstep = (float*)d_out + cB * cCW + (size_t)step * cB * cV;
    k_mask<<<cB, 256, 0, stream>>>(tokens, logits, disallowed, outstep, mwork, pos);
    k_topk_sample<<<cB, 256, 0, stream>>>(mwork, disallowed, tokens, pos, step);
  }
  k_out_tokens<<<(cB * cCW + 255) / 256, 256, 0, stream>>>(tokens, (int*)d_out);
}
